// VectorQuantizer_25159918420816
// MI455X (gfx1250) — matrix operand order repaired and checked by reading, not compiled, not run on hardware
//
#include <hip/hip_runtime.h>
#include <hip/hip_bf16.h>

typedef __attribute__((ext_vector_type(16))) __bf16 v16bf;
typedef __attribute__((ext_vector_type(8)))  __bf16 v8bf;
typedef __attribute__((ext_vector_type(8)))  float  v8f;
typedef unsigned int u32x4 __attribute__((ext_vector_type(4)));
typedef int          i32x4 __attribute__((ext_vector_type(4)));
typedef int          i32x8 __attribute__((ext_vector_type(8)));

#define C_DIM   256
#define K_CODES 1024
#define THW     8192
#define N_PTS   32768
#define BROW    264   // LDS row stride in bf16: 512B data + 16B TDM pad = 528B (4-DW skew)

// ---------------------------------------------------------------------------
// TDM: load one 16-row x 256-col bf16 codebook tile (2D tensor descriptor) to
// LDS with row padding (pad_interval=128 DWORDs, pad_amount=4 DWORDs) so rows
// land on distinct LDS bank phases. Issued wave-level; tracked by TENSORcnt.
// D# layout per CDNA5 ISA.
// ---------------------------------------------------------------------------
__device__ __forceinline__ void tdm_load_tile(const __bf16* gsrc, unsigned lds_byte) {
    unsigned long long ga = (unsigned long long)(const void*)gsrc;
    u32x4 g0;
    g0[0] = 1u;                                          // count=1, user descriptor
    g0[1] = lds_byte;                                    // lds_addr (bytes)
    g0[2] = (unsigned)(ga & 0xFFFFFFFFu);                // global_addr[31:0]
    g0[3] = (unsigned)((ga >> 32) & 0x01FFFFFFu)         // global_addr[56:32]
          | 0x80000000u;                                 // type=2 ("image")
    i32x8 g1;
    g1[0] = (1 << 16)                                    // data_size = 2 bytes
          | (1 << 20)                                    // pad_enable
          | (6 << 22)                                    // pad_interval: 128 DWORDs (one row)
          | (3 << 25);                                   // pad_amount: 4 DWORDs (16B)
    g1[1] = (C_DIM & 0xFFFF) << 16;                      // tensor_dim0 = 256
    g1[2] = (K_CODES & 0xFFFF) << 16;                    // tensor_dim1 = 1024
    g1[3] = (C_DIM << 16);                               // tile_dim0 = 256
    g1[4] = 16;                                          // tile_dim1 = 16, tile_dim2 = 0
    g1[5] = C_DIM;                                       // tensor_dim0_stride = 256
    g1[6] = 0;
    g1[7] = 0;
    i32x4 gz = {0, 0, 0, 0};
#if defined(__clang_major__) && (__clang_major__ >= 23)
    i32x8 gz8 = {0, 0, 0, 0, 0, 0, 0, 0};
    __builtin_amdgcn_tensor_load_to_lds(g0, g1, gz, gz, gz8, 0);
#else
    __builtin_amdgcn_tensor_load_to_lds(g0, g1, gz, gz, 0);
#endif
}

// Flat shared-memory address -> LDS byte offset (HW maps LDS aperture by addr[31:0]).
__device__ __forceinline__ unsigned lds_offset(const void* p) {
    return (unsigned)(unsigned long long)p;
}

// ---------------------------------------------------------------------------
// Kernel 1: codebook prep: fp32 -> bf16 (row major, K x C) + exact fp32 norms.
// ---------------------------------------------------------------------------
__global__ void __launch_bounds__(32)
emb_prep_kernel(const float* __restrict__ emb, __bf16* __restrict__ ebf,
                float* __restrict__ enorm) {
    const int k = blockIdx.x;
    const int lane = threadIdx.x;
    const float* row = emb + k * C_DIM + lane * 8;
    float4 a = *(const float4*)(row);
    float4 b = *(const float4*)(row + 4);
    float s = a.x * a.x + a.y * a.y + a.z * a.z + a.w * a.w
            + b.x * b.x + b.y * b.y + b.z * b.z + b.w * b.w;
    v8bf o;
    o[0] = (__bf16)a.x; o[1] = (__bf16)a.y; o[2] = (__bf16)a.z; o[3] = (__bf16)a.w;
    o[4] = (__bf16)b.x; o[5] = (__bf16)b.y; o[6] = (__bf16)b.z; o[7] = (__bf16)b.w;
    *(v8bf*)(ebf + k * C_DIM + lane * 8) = o;
#pragma unroll
    for (int off = 16; off > 0; off >>= 1) s += __shfl_xor(s, off, 32);
    if (lane == 0) enorm[k] = s;
}

// ---------------------------------------------------------------------------
// Kernel 2: z (B,C,T,H,W) fp32 -> zt (N,C) bf16, N = b*THW + thw (LDS tiled).
// ---------------------------------------------------------------------------
__global__ void __launch_bounds__(256)
z_transpose_kernel(const float* __restrict__ z, __bf16* __restrict__ zt) {
    __shared__ __bf16 tile[64][258];  // 258*2B/row = 129 DW (odd) -> conflict-free
    const int b    = blockIdx.x >> 7;
    const int thw0 = (blockIdx.x & 127) << 6;
    const float* zb = z + (size_t)b * C_DIM * THW + thw0;
    for (int e = threadIdx.x; e < 64 * C_DIM; e += 256) {
        const int c = e >> 6, tl = e & 63;
        tile[tl][c] = (__bf16)zb[(size_t)c * THW + tl];
    }
    __syncthreads();
    unsigned* ztw = (unsigned*)zt;
    const int n0 = b * THW + thw0;
    for (int e = threadIdx.x; e < 64 * 128; e += 256) {
        const int r = e >> 7, col = e & 127;
        ztw[(size_t)(n0 + r) * 128 + col] = *(const unsigned*)&tile[r][col * 2];
    }
}

// ---------------------------------------------------------------------------
// Kernel 3: WMMA distance sweep + running argmin, B staged via TDM double-buffer.
// Block = 256 threads = 8 waves; each wave owns a 16-row (n) tile; all waves
// share the LDS-resident 16-code B tile. score = ||e_k||^2 - 2 z.e_k.
// ---------------------------------------------------------------------------
__global__ void __launch_bounds__(256)
vq_argmin_kernel(const __bf16* __restrict__ zt, const __bf16* __restrict__ ebf,
                 const float* __restrict__ enorm, int* __restrict__ idx_out,
                 float* __restrict__ idxf_out) {
    __shared__ __bf16 bbuf[2][16 * BROW];   // 2 x 8.25 KB double-buffered B tiles
    __shared__ float  snorm[K_CODES];       // 4 KB
    __shared__ float  sval[8][32][8];
    __shared__ int    sidx[8][32][8];

    const int tid  = threadIdx.x;
    const int wave = tid >> 5, lane = tid & 31;
    const int lrow = lane & 15, lh = lane >> 4;
    const int n0   = (blockIdx.x * 8 + wave) * 16;

    // Preload code norms into LDS.
    for (int e = tid; e < K_CODES; e += 256) snorm[e] = enorm[e];

    // Cache full A tile (16 x 256 bf16). ISA A-layout (16-bit, 16x32):
    // lane half h holds K=[h*8,h*8+8) in VGPR0-3 and K=[16+h*8,...) in VGPR4-7.
    const __bf16* arow = zt + (size_t)(n0 + lrow) * C_DIM;
    v16bf A[8];
#pragma unroll
    for (int ci = 0; ci < 8; ++ci) {
        v8bf lo = *(const v8bf*)(arow + ci * 32 + lh * 8);
        v8bf hi = *(const v8bf*)(arow + ci * 32 + 16 + lh * 8);
        A[ci] = __builtin_shufflevector(lo, hi, 0, 1, 2, 3, 4, 5, 6, 7,
                                        8, 9, 10, 11, 12, 13, 14, 15);
    }

    const unsigned lds_b[2] = { lds_offset(&bbuf[0][0]), lds_offset(&bbuf[1][0]) };

    // Prologue: TDM fetch of tile 0 (wave 0 only; EXEC-ignored wave-level op).
    if (wave == 0) {
        tdm_load_tile(ebf, lds_b[0]);
        __builtin_amdgcn_s_wait_tensorcnt(0);
    }
    __syncthreads();

    float minv[8];
    int   mini[8];
#pragma unroll
    for (int r = 0; r < 8; ++r) { minv[r] = 3.4e38f; mini[r] = 0; }

    for (int t = 0; t < K_CODES / 16; ++t) {
        // Kick off DMA of the next B tile while this one is consumed.
        if (wave == 0 && t + 1 < K_CODES / 16)
            tdm_load_tile(ebf + (size_t)(t + 1) * 16 * C_DIM, lds_b[(t + 1) & 1]);

        // B-layout (32x16 bf16): lane = column (code), half h holds K=[h*8,h*8+8)
        // and K=[16+h*8,16+h*8+8): one padded LDS row per code.
        const __bf16* bl = &bbuf[t & 1][lrow * BROW + lh * 8];
        v8f acc = {0.f, 0.f, 0.f, 0.f, 0.f, 0.f, 0.f, 0.f};
#pragma unroll
        for (int ci = 0; ci < 8; ++ci) {
            v8bf lo = *(const v8bf*)(bl + ci * 32);
            v8bf hi = *(const v8bf*)(bl + ci * 32 + 16);
            v16bf Bc = __builtin_shufflevector(lo, hi, 0, 1, 2, 3, 4, 5, 6, 7,
                                               8, 9, 10, 11, 12, 13, 14, 15);
            acc = __builtin_amdgcn_wmma_f32_16x16x32_bf16(
                false, A[ci], false, Bc, (short)0, acc, false, false);
        }
        const int   kk = t * 16 + lrow;      // column this lane holds (both halves)
        const float en = snorm[kk];
#pragma unroll
        for (int r = 0; r < 8; ++r) {        // C/D layout: VGPR r -> row r (+8 upper half)
            const float s = en - 2.0f * acc[r];
            if (s < minv[r]) { minv[r] = s; mini[r] = kk; }
        }

        // Next buffer must be complete before anyone reads it; the barrier also
        // guarantees all waves finished reading buf before the DMA-after-next.
        if (wave == 0) __builtin_amdgcn_s_wait_tensorcnt(0);
        __syncthreads();
    }

#pragma unroll
    for (int r = 0; r < 8; ++r) {
        sval[wave][lane][r] = minv[r];
        sidx[wave][lane][r] = mini[r];
    }
    __syncthreads();
    if (lane < 16) {
        const int M = lane;
        const int base = (M < 8) ? 0 : 16;
        const int r = M & 7;
        float best = 3.4e38f;
        int   bidx = 0x7fffffff;
#pragma unroll
        for (int j = 0; j < 16; ++j) {
            const float v  = sval[wave][base + j][r];
            const int   ix = sidx[wave][base + j][r];
            if (v < best || (v == best && ix < bidx)) { best = v; bidx = ix; }
        }
        idx_out[n0 + M]  = bidx;
        idxf_out[n0 + M] = (float)bidx;
    }
}

// ---------------------------------------------------------------------------
// Kernel 4: gather z_q = emb[idx] scattered back to (B,C,T,H,W).
// ---------------------------------------------------------------------------
__global__ void __launch_bounds__(256)
vq_gather_kernel(const float* __restrict__ emb, const int* __restrict__ idx,
                 float* __restrict__ zq) {
    const int n = blockIdx.x * 256 + threadIdx.x;
    const int b = n >> 13, thw = n & (THW - 1);
    const float* row = emb + (size_t)idx[n] * C_DIM;
    float* outp = zq + (size_t)b * C_DIM * THW + thw;
#pragma unroll 8
    for (int c = 0; c < C_DIM; c += 4) {
        const float4 v = *(const float4*)(row + c);
        outp[(size_t)(c + 0) * THW] = v.x;
        outp[(size_t)(c + 1) * THW] = v.y;
        outp[(size_t)(c + 2) * THW] = v.z;
        outp[(size_t)(c + 3) * THW] = v.w;
    }
}

// ---------------------------------------------------------------------------
extern "C" void kernel_launch(void* const* d_in, const int* in_sizes, int n_in,
                              void* d_out, int out_size, void* d_ws, size_t ws_size,
                              hipStream_t stream) {
    (void)in_sizes; (void)n_in; (void)out_size; (void)ws_size;
    const float* z   = (const float*)d_in[0];   // (4,256,8,32,32) fp32
    const float* emb = (const float*)d_in[1];   // (1024,256) fp32

    float* out  = (float*)d_out;
    float* zq   = out;                 // 8,388,608 floats
    float* idxf = out + 8388608;       // 32,768 floats (indices as fp32)

    // Stage transposed bf16 z inside the z_q output region (16 MB < 33.5 MB);
    // fully overwritten by the gather kernel afterwards. Keeps ws use small.
    __bf16* zt = (__bf16*)d_out;

    char* ws = (char*)d_ws;
    __bf16* ebf   = (__bf16*)ws;                       // 512 KB
    float*  enorm = (float*)(ws + 524288);             // 4 KB
    int*    idx   = (int*)(ws + 524288 + 4096);        // 128 KB

    emb_prep_kernel   <<<K_CODES, 32, 0, stream>>>(emb, ebf, enorm);
    z_transpose_kernel<<<512, 256, 0, stream>>>(z, zt);
    vq_argmin_kernel  <<<256, 256, 0, stream>>>(zt, ebf, enorm, idx, idxf);
    vq_gather_kernel  <<<128, 256, 0, stream>>>(emb, idx, zq);
}
